// Module3_24910810317504
// MI455X (gfx1250) — matrix operand order repaired and checked by reading, not compiled, not run on hardware
//
#include <hip/hip_runtime.h>

// ============================================================================
// Hierarchical attention network on gfx1250 (MI455X).
// All GEMMs run through V_WMMA_F32_16X16X32_F16 (f16 operands, f32 accum).
// GEMM: software-pipelined K loop (register prefetch of next chunk overlaps
// WMMA, global_prefetch_b8 warms GL2 two chunks ahead), branch-free clamped
// A staging, batched b128 fragment loads, back-to-back WMMAs, pointer-bump
// addressing (no 64-bit multiplies in the loop).
// ============================================================================

typedef _Float16 f16;
typedef _Float16 half16 __attribute__((ext_vector_type(16)));
typedef float    float8 __attribute__((ext_vector_type(8)));

// ---- model dims ----
#define NSENT   2560   // total sentences
#define SLEN    40     // tokens per sentence
#define NDOC    128    // docs
#define DTRUNC  20     // sentences per doc
#define NBLOG   8
#define BTRUNC  16     // docs per blog
#define DIM     256    // embedding dim
#define HGRU    256    // GRU hidden
#define G3      768    // 3*H
#define H2      512    // 2*H
#define NPOS    50

// ---------------------------------------------------------------------------
// WMMA GEMM: C[M,N] = A[M,K] * B[K,N] + bias[N]; A,B f16 row-major; C f32/f16.
// 256 threads = 8 waves. Block tile 128x64, K chunk 32.
// Wave w computes rows [16w,16w+16) x 64 cols (4 16x16 f32 accumulators).
// Fragment layouts (CDNA5 ISA 7.12.2, wave32):
//   A 16x32 f16 : lanes 0-15 row=lane, halves 0-7 -> K 0-7, 8-15 -> K 16-23;
//                 lanes 16-31 row=lane-16, K 8-15 / K 24-31.
//                 -> per lane: two contiguous 16B spans of the As row.
//   B 32x16 f16 : lanes 0-15 col=lane, halves 0-7 -> K 0-7, 8-15 -> K 16-23; lanes 16-31 K 8-15 / K 24-31.
//                 -> per lane: two contiguous 16B spans of the BsT row.
//   C 16x16 f32 : vgpr r, lanes 0-15 (M=r,N=lane), lanes 16-31 (M=8+r,N=l-16).
// ---------------------------------------------------------------------------
#define BM 128
#define BN 64
#define KC 32

__global__ __launch_bounds__(256) void wmma_gemm_kernel(
    const f16* __restrict__ A, int lda,
    const f16* __restrict__ B, int ldb,
    const float* __restrict__ bias,
    float* __restrict__ Cf, f16* __restrict__ Ch,
    int M, int K, int N)
{
  __shared__ __align__(16) f16 As[BM][KC];   // row-major tile, row = 64B
  __shared__ __align__(16) f16 BsT[BN][KC];  // transposed tile: [n][k], row = 64B

  const int tid  = threadIdx.x;
  const int wave = tid >> 5;    // 0..7
  const int lane = tid & 31;
  const int r16  = lane & 15;
  const int hi   = lane >> 4;

  const int rowBase = blockIdx.x * BM;
  const int colBase = blockIdx.y * BN;

  // ---- per-thread staging coordinates (constant across K chunks) ----
  int aLdsRow[4], aCol[4];
  const f16* aPtr[4];
  #pragma unroll
  for (int i = 0; i < 4; ++i) {
    int chunk = tid + i * 256;              // 1024 chunks of 4 halves (A)
    int r = chunk >> 3;                     // 0..127
    aLdsRow[i] = r;
    aCol[i] = (chunk & 7) << 2;             // 0..28
    int gr = rowBase + r;
    if (gr > M - 1) gr = M - 1;             // clamp: branch-free, always in-bounds
    aPtr[i] = A + (size_t)gr * lda + aCol[i];
  }
  int bRow[2], bCol[2];
  const f16* bPtr[2];
  #pragma unroll
  for (int i = 0; i < 2; ++i) {
    int chunk = tid + i * 256;              // 512 chunks of 4 halves (B)
    bRow[i] = chunk >> 4;                   // k: 0..31
    bCol[i] = (chunk & 15) << 2;            // n: 0..60
    bPtr[i] = B + (size_t)bRow[i] * ldb + (colBase + bCol[i]);
  }
  const size_t bStep = (size_t)KC * ldb;    // pointer bump per K chunk

  uint2 ra[4], rb[2];
  auto prefetch = [&]() {                   // load current chunk, bump pointers
    #pragma unroll
    for (int i = 0; i < 4; ++i) {
      ra[i] = *reinterpret_cast<const uint2*>(aPtr[i]);
      aPtr[i] += KC;
    }
    #pragma unroll
    for (int i = 0; i < 2; ++i) {
      rb[i] = *reinterpret_cast<const uint2*>(bPtr[i]);
      bPtr[i] += bStep;
    }
  };

  float8 acc[4] = {};

  prefetch();                               // chunk 0 -> registers
  for (int k0 = 0; k0 < K; k0 += KC) {
    // ---- commit prefetched tile to LDS ----
    #pragma unroll
    for (int i = 0; i < 4; ++i)
      *reinterpret_cast<uint2*>(&As[aLdsRow[i]][aCol[i]]) = ra[i];
    #pragma unroll
    for (int i = 0; i < 2; ++i) {
      union { uint2 u; f16 h[4]; } v;
      v.u = rb[i];
      BsT[bCol[i] + 0][bRow[i]] = v.h[0];
      BsT[bCol[i] + 1][bRow[i]] = v.h[1];
      BsT[bCol[i] + 2][bRow[i]] = v.h[2];
      BsT[bCol[i] + 3][bRow[i]] = v.h[3];
    }
    __syncthreads();

    // ---- pipeline: next chunk -> registers, chunk+2 -> GL2 prefetch ----
    if (k0 + KC < K) {
      prefetch();
      if (k0 + 2 * KC < K) {
        __builtin_prefetch(aPtr[0], 0, 1);  // global_prefetch_b8
        __builtin_prefetch(bPtr[0], 0, 1);
      }
    }

    // ---- batched fragment loads: 10x ds_load_b128 ----
    union { half16 v; uint4 q[2]; } af;
    const uint4* pA = reinterpret_cast<const uint4*>(&As[wave * 16 + r16][0]);
    af.q[0] = pA[hi];        // K k0+hi*8 .. +7
    af.q[1] = pA[2 + hi];    // K k0+16+hi*8 .. +7

    union { half16 v; uint4 q[2]; } bf[4];
    #pragma unroll
    for (int nt = 0; nt < 4; ++nt) {
      const uint4* pB = reinterpret_cast<const uint4*>(&BsT[nt * 16 + r16][0]);
      bf[nt].q[0] = pB[hi];
      bf[nt].q[1] = pB[2 + hi];
    }

    // ---- 4 back-to-back WMMAs (independent accumulators, no RAW hazard) ----
    #pragma unroll
    for (int nt = 0; nt < 4; ++nt)
      acc[nt] = __builtin_amdgcn_wmma_f32_16x16x32_f16(
          false, af.v, false, bf[nt].v, (short)0, acc[nt], false, false);

    __syncthreads();
  }

  // ---- store: uniform fast path when the whole block tile is in range ----
  const bool fullM = (rowBase + BM <= M);
  if (Cf) {
    #pragma unroll
    for (int nt = 0; nt < 4; ++nt) {
      const int gc = colBase + nt * 16 + r16;
      const float badd = bias ? bias[gc] : 0.0f;
      #pragma unroll
      for (int r = 0; r < 8; ++r) {
        const int gr = rowBase + wave * 16 + hi * 8 + r;
        if (fullM || gr < M) Cf[(size_t)gr * N + gc] = acc[nt][r] + badd;
      }
    }
  } else {
    #pragma unroll
    for (int nt = 0; nt < 4; ++nt) {
      const int gc = colBase + nt * 16 + r16;
      const float badd = bias ? bias[gc] : 0.0f;
      #pragma unroll
      for (int r = 0; r < 8; ++r) {
        const int gr = rowBase + wave * 16 + hi * 8 + r;
        if (fullM || gr < M) Ch[(size_t)gr * N + gc] = (f16)(acc[nt][r] + badd);
      }
    }
  }
}

// ---------------------------------------------------------------------------
// helpers
// ---------------------------------------------------------------------------
__global__ void cvt_f16_kernel(const float* __restrict__ s, f16* __restrict__ d, int n) {
  int i = blockIdx.x * 256 + threadIdx.x;
  if (i < n) d[i] = (f16)s[i];
}

// one block (=DIM threads) per token
__global__ void gather_kernel(const int* __restrict__ x, const float* __restrict__ embed,
                              f16* __restrict__ xe, int ntok) {
  int tok = blockIdx.x;
  if (tok < ntok)
    xe[(size_t)tok * DIM + threadIdx.x] =
        (f16)embed[(size_t)x[tok] * DIM + threadIdx.x];
}

__global__ void zero_state_kernel(float* __restrict__ h, f16* __restrict__ hh, int n) {
  int i = blockIdx.x * 256 + threadIdx.x;
  if (i < n) { h[i] = 0.0f; hh[i] = (f16)0.0f; }
}

// GRU cell: r/z/n gates, in-place f32 state update + f16 copy + sequence write.
// xi layout [Bt, T, 768] (f16, bias folded); gh [Bt, 768] f32 (bh folded);
// ys layout [Bt, T, 512], this direction writes cols [col_off, col_off+256).
__global__ void gru_step_kernel(
    const f16* __restrict__ xi, int T, int t_xi,
    const float* __restrict__ gh,
    float* __restrict__ h, f16* __restrict__ hh,
    float* __restrict__ ys, int t_out, int col_off, int Bt)
{
  int idx = blockIdx.x * 256 + threadIdx.x;
  if (idx >= Bt * HGRU) return;
  int b = idx >> 8;          // HGRU == 256
  int j = idx & 255;
  const f16*   xr = xi + ((size_t)b * T + t_xi) * G3;
  const float* gr = gh + (size_t)b * G3;
  float x_r = (float)xr[j],        g_r = gr[j];
  float x_z = (float)xr[256 + j],  g_z = gr[256 + j];
  float x_n = (float)xr[512 + j],  g_n = gr[512 + j];
  float r = 1.0f / (1.0f + __expf(-(x_r + g_r)));
  float z = 1.0f / (1.0f + __expf(-(x_z + g_z)));
  float n = tanhf(x_n + r * g_n);
  float hn = (1.0f - z) * n + z * h[idx];
  h[idx]  = hn;
  hh[idx] = (f16)hn;
  ys[((size_t)b * T + t_out) * H2 + col_off + j] = hn;
}

// Attention pooling: one block per batch row. ctx [B,T,512], query [512].
// (Masks are all-false in this problem: tokens are 1..V-1 and all docs full.)
__global__ __launch_bounds__(256) void attn_kernel(
    const float* __restrict__ ctx, const float* __restrict__ query,
    float* __restrict__ out, int T)
{
  __shared__ float sc[64];
  int b = blockIdx.x, tid = threadIdx.x;
  int wave = tid >> 5, lane = tid & 31;
  for (int t = wave; t < T; t += 8) {
    const float* row = ctx + ((size_t)b * T + t) * H2;
    float p = 0.0f;
    for (int d = lane; d < H2; d += 32) p += row[d] * query[d];
    #pragma unroll
    for (int off = 16; off > 0; off >>= 1) p += __shfl_xor(p, off, 32);
    if (lane == 0) sc[t] = p;
  }
  __syncthreads();
  if (tid == 0) {
    float m = -1e30f;
    for (int t = 0; t < T; ++t) m = fmaxf(m, sc[t]);
    float s = 0.0f;
    for (int t = 0; t < T; ++t) { float e = __expf(sc[t] - m); sc[t] = e; s += e; }
    float inv = 1.0f / s;
    for (int t = 0; t < T; ++t) sc[t] *= inv;
  }
  __syncthreads();
  for (int d = tid; d < H2; d += 256) {
    float a = 0.0f;
    for (int t = 0; t < T; ++t) a += sc[t] * ctx[((size_t)b * T + t) * H2 + d];
    out[(size_t)b * H2 + d] = a;
  }
}

// doc head: sigmoid(dv.cw + dv^T W bv + pos + bias); one block per doc.
__global__ __launch_bounds__(256) void doc_prob_kernel(
    const float* __restrict__ doc_vec, const float* __restrict__ blog_vec,
    const float* __restrict__ content_w, const float* __restrict__ sal_w,
    const float* __restrict__ pos_embed, const float* __restrict__ pos_w,
    const float* __restrict__ bias, float* __restrict__ out)
{
  __shared__ float dv[H2], bv[H2], red[256];
  int n = blockIdx.x, tid = threadIdx.x;
  int b = n >> 4, j = n & 15;
  for (int i = tid; i < H2; i += 256) {
    dv[i] = doc_vec[(size_t)n * H2 + i];
    bv[i] = blog_vec[(size_t)b * H2 + i];
  }
  __syncthreads();
  float part = 0.0f;
  for (int e = tid; e < H2; e += 256) {
    float t = 0.0f;
    for (int d = 0; d < H2; ++d) t += dv[d] * sal_w[(size_t)d * H2 + e];
    part += t * bv[e];
    part += dv[e] * content_w[e];
  }
  for (int p = tid; p < NPOS; p += 256) part += pos_embed[j * NPOS + p] * pos_w[p];
  red[tid] = part; __syncthreads();
  for (int s = 128; s > 0; s >>= 1) { if (tid < s) red[tid] += red[tid + s]; __syncthreads(); }
  if (tid == 0) out[n] = 1.0f / (1.0f + __expf(-(red[0] + bias[0])));
}

// sentence head; preserves reference bug: doc label uses doc_probs[blog_idx].
__global__ __launch_bounds__(256) void sent_prob_kernel(
    const float* __restrict__ sent_vec, const float* __restrict__ doc_vec,
    const float* __restrict__ blog_vec,
    const float* __restrict__ content_w, const float* __restrict__ sal_w, // [512,1024]
    const float* __restrict__ pos_embed, const float* __restrict__ pos_w,
    const float* __restrict__ dlabel_w, const float* __restrict__ bias,
    const float* __restrict__ doc_probs, float* __restrict__ out)
{
  __shared__ float sv[H2], ctx[2 * H2], red[256];
  int n = blockIdx.x, tid = threadIdx.x;
  int dn = n / DTRUNC, k = n % DTRUNC;
  int bn = dn >> 4;
  for (int i = tid; i < H2; i += 256) {
    sv[i]       = sent_vec[(size_t)n * H2 + i];
    ctx[i]      = blog_vec[(size_t)bn * H2 + i];
    ctx[H2 + i] = doc_vec[(size_t)dn * H2 + i];
  }
  __syncthreads();
  float part = 0.0f;
  for (int e = tid; e < 2 * H2; e += 256) {
    float t = 0.0f;
    for (int d = 0; d < H2; ++d) t += sv[d] * sal_w[(size_t)d * 1024 + e];
    part += t * ctx[e];
  }
  for (int d = tid; d < H2; d += 256) part += sv[d] * content_w[d];
  for (int p = tid; p < NPOS; p += 256) part += pos_embed[k * NPOS + p] * pos_w[p];
  red[tid] = part; __syncthreads();
  for (int s = 128; s > 0; s >>= 1) { if (tid < s) red[tid] += red[tid + s]; __syncthreads(); }
  if (tid == 0)
    out[n] = 1.0f / (1.0f + __expf(-(red[0] + dlabel_w[0] * doc_probs[bn] + bias[0])));
}

// ---------------------------------------------------------------------------
// Host orchestration. Workspace footprint ~620 MB.
// d_in order (setup_inputs dict, gru dicts flattened Wi,Wh,bi,bh):
//  0 x  1 doc_nums  2 doc_lens  3 embed
//  4-7 word_f  8-11 word_b  12-15 sent_f  16-19 sent_b  20-23 doc_f  24-27 doc_b
//  28 word_query 29 sent_query 30 doc_query
//  31 doc_content_w 32 doc_salience_w 33 doc_pos_embed 34 doc_pos_w 35 doc_bias
//  36 sent_content_w 37 sent_salience_w 38 sent_pos_embed 39 sent_pos_w
//  40 sent_doc_label_w 41 sent_bias
// ---------------------------------------------------------------------------
extern "C" void kernel_launch(void* const* d_in, const int* in_sizes, int n_in,
                              void* d_out, int out_size, void* d_ws, size_t ws_size,
                              hipStream_t stream)
{
  (void)in_sizes; (void)n_in; (void)out_size; (void)ws_size;

  const int*   x      = (const int*)  d_in[0];
  const float* embed  = (const float*)d_in[3];
  const float* wf_Wi = (const float*)d_in[4],  *wf_Wh = (const float*)d_in[5];
  const float* wf_bi = (const float*)d_in[6],  *wf_bh = (const float*)d_in[7];
  const float* wb_Wi = (const float*)d_in[8],  *wb_Wh = (const float*)d_in[9];
  const float* wb_bi = (const float*)d_in[10], *wb_bh = (const float*)d_in[11];
  const float* sf_Wi = (const float*)d_in[12], *sf_Wh = (const float*)d_in[13];
  const float* sf_bi = (const float*)d_in[14], *sf_bh = (const float*)d_in[15];
  const float* sb_Wi = (const float*)d_in[16], *sb_Wh = (const float*)d_in[17];
  const float* sb_bi = (const float*)d_in[18], *sb_bh = (const float*)d_in[19];
  const float* df_Wi = (const float*)d_in[20], *df_Wh = (const float*)d_in[21];
  const float* df_bi = (const float*)d_in[22], *df_bh = (const float*)d_in[23];
  const float* db_Wi = (const float*)d_in[24], *db_Wh = (const float*)d_in[25];
  const float* db_bi = (const float*)d_in[26], *db_bh = (const float*)d_in[27];
  const float* word_q = (const float*)d_in[28];
  const float* sent_q = (const float*)d_in[29];
  const float* doc_q  = (const float*)d_in[30];
  const float* d_cw = (const float*)d_in[31], *d_sw = (const float*)d_in[32];
  const float* d_pe = (const float*)d_in[33], *d_pw = (const float*)d_in[34];
  const float* d_b  = (const float*)d_in[35];
  const float* s_cw = (const float*)d_in[36], *s_sw = (const float*)d_in[37];
  const float* s_pe = (const float*)d_in[38], *s_pw = (const float*)d_in[39];
  const float* s_dlw = (const float*)d_in[40], *s_b = (const float*)d_in[41];

  // bump allocator over d_ws
  char* base = (char*)d_ws;
  size_t off = 0;
  auto alloc = [&](size_t bytes) -> void* {
    off = (off + 255) & ~(size_t)255;
    void* p = base + off;
    off += bytes;
    return p;
  };

  // f16 weight copies
  f16* wWiF = (f16*)alloc((size_t)DIM  * G3 * 2);
  f16* wWhF = (f16*)alloc((size_t)HGRU * G3 * 2);
  f16* wWiB = (f16*)alloc((size_t)DIM  * G3 * 2);
  f16* wWhB = (f16*)alloc((size_t)HGRU * G3 * 2);
  f16* sWiF = (f16*)alloc((size_t)H2   * G3 * 2);
  f16* sWhF = (f16*)alloc((size_t)HGRU * G3 * 2);
  f16* sWiB = (f16*)alloc((size_t)H2   * G3 * 2);
  f16* sWhB = (f16*)alloc((size_t)HGRU * G3 * 2);
  f16* dWiF = (f16*)alloc((size_t)H2   * G3 * 2);
  f16* dWhF = (f16*)alloc((size_t)HGRU * G3 * 2);
  f16* dWiB = (f16*)alloc((size_t)H2   * G3 * 2);
  f16* dWhB = (f16*)alloc((size_t)HGRU * G3 * 2);

  // activations
  f16*   xe       = (f16*)  alloc((size_t)NSENT * SLEN * DIM * 2);
  f16*   xiWf     = (f16*)  alloc((size_t)NSENT * SLEN * G3 * 2);
  f16*   xiWb     = (f16*)  alloc((size_t)NSENT * SLEN * G3 * 2);
  float* wh       = (float*)alloc((size_t)NSENT * SLEN * H2 * 4);
  float* hbuf     = (float*)alloc((size_t)NSENT * HGRU * 4);
  f16*   hbuf_h   = (f16*)  alloc((size_t)NSENT * HGRU * 2);
  float* ghbuf    = (float*)alloc((size_t)NSENT * G3 * 4);
  float* sent_vec = (float*)alloc((size_t)NSENT * H2 * 4);
  f16*   sent_h   = (f16*)  alloc((size_t)NSENT * H2 * 2);
  f16*   xiSf     = (f16*)  alloc((size_t)NSENT * G3 * 2);
  f16*   xiSb     = (f16*)  alloc((size_t)NSENT * G3 * 2);
  float* sh       = (float*)alloc((size_t)NDOC * DTRUNC * H2 * 4);
  float* doc_vec  = (float*)alloc((size_t)NDOC * H2 * 4);
  f16*   doc_h    = (f16*)  alloc((size_t)NDOC * H2 * 2);
  f16*   xiDf     = (f16*)  alloc((size_t)NDOC * G3 * 2);
  f16*   xiDb     = (f16*)  alloc((size_t)NDOC * G3 * 2);
  float* dh       = (float*)alloc((size_t)NBLOG * BTRUNC * H2 * 4);
  float* blog_vec = (float*)alloc((size_t)NBLOG * H2 * 4);

  auto cvt = [&](const float* s, f16* d, int n) {
    cvt_f16_kernel<<<(n + 255) / 256, 256, 0, stream>>>(s, d, n);
  };
  auto gemm = [&](const f16* A, int lda, const f16* B, int ldb, const float* bias,
                  float* Cf, f16* Ch, int M, int K, int N) {
    dim3 g((M + BM - 1) / BM, N / BN);
    wmma_gemm_kernel<<<g, 256, 0, stream>>>(A, lda, B, ldb, bias, Cf, Ch, M, K, N);
  };
  auto scan = [&](int Bt, int T, const f16* xi, const f16* Wh_h, const float* bh,
                  float* ys, int coloff, bool rev) {
    int n = Bt * HGRU;
    zero_state_kernel<<<(n + 255) / 256, 256, 0, stream>>>(hbuf, hbuf_h, n);
    for (int s = 0; s < T; ++s) {
      gemm(hbuf_h, HGRU, Wh_h, G3, bh, ghbuf, nullptr, Bt, HGRU, G3);
      int t = rev ? (T - 1 - s) : s;
      gru_step_kernel<<<(n + 255) / 256, 256, 0, stream>>>(
          xi, T, t, ghbuf, hbuf, hbuf_h, ys, t, coloff, Bt);
    }
  };

  // ---- weights -> f16 ----
  cvt(wf_Wi, wWiF, DIM * G3);  cvt(wf_Wh, wWhF, HGRU * G3);
  cvt(wb_Wi, wWiB, DIM * G3);  cvt(wb_Wh, wWhB, HGRU * G3);
  cvt(sf_Wi, sWiF, H2 * G3);   cvt(sf_Wh, sWhF, HGRU * G3);
  cvt(sb_Wi, sWiB, H2 * G3);   cvt(sb_Wh, sWhB, HGRU * G3);
  cvt(df_Wi, dWiF, H2 * G3);   cvt(df_Wh, dWhF, HGRU * G3);
  cvt(db_Wi, dWiB, H2 * G3);   cvt(db_Wh, dWhB, HGRU * G3);

  // ---- word level ----
  gather_kernel<<<NSENT * SLEN, DIM, 0, stream>>>(x, embed, xe, NSENT * SLEN);
  gemm(xe, DIM, wWiF, G3, wf_bi, nullptr, xiWf, NSENT * SLEN, DIM, G3);
  gemm(xe, DIM, wWiB, G3, wb_bi, nullptr, xiWb, NSENT * SLEN, DIM, G3);
  scan(NSENT, SLEN, xiWf, wWhF, wf_bh, wh, 0,    false);
  scan(NSENT, SLEN, xiWb, wWhB, wb_bh, wh, HGRU, true);
  attn_kernel<<<NSENT, 256, 0, stream>>>(wh, word_q, sent_vec, SLEN);

  // ---- sentence level ----
  cvt(sent_vec, sent_h, NSENT * H2);
  gemm(sent_h, H2, sWiF, G3, sf_bi, nullptr, xiSf, NSENT, H2, G3);
  gemm(sent_h, H2, sWiB, G3, sb_bi, nullptr, xiSb, NSENT, H2, G3);
  scan(NDOC, DTRUNC, xiSf, sWhF, sf_bh, sh, 0,    false);
  scan(NDOC, DTRUNC, xiSb, sWhB, sb_bh, sh, HGRU, true);
  attn_kernel<<<NDOC, 256, 0, stream>>>(sh, sent_q, doc_vec, DTRUNC);

  // ---- doc level ----
  cvt(doc_vec, doc_h, NDOC * H2);
  gemm(doc_h, H2, dWiF, G3, df_bi, nullptr, xiDf, NDOC, H2, G3);
  gemm(doc_h, H2, dWiB, G3, db_bi, nullptr, xiDb, NDOC, H2, G3);
  scan(NBLOG, BTRUNC, xiDf, dWhF, df_bh, dh, 0,    false);
  scan(NBLOG, BTRUNC, xiDb, dWhB, db_bh, dh, HGRU, true);
  attn_kernel<<<NBLOG, 256, 0, stream>>>(dh, doc_q, blog_vec, BTRUNC);

  // ---- heads ----
  float* out = (float*)d_out;
  doc_prob_kernel<<<NDOC, 256, 0, stream>>>(
      doc_vec, blog_vec, d_cw, d_sw, d_pe, d_pw, d_b, out);
  sent_prob_kernel<<<NSENT, 256, 0, stream>>>(
      sent_vec, doc_vec, blog_vec, s_cw, s_sw, s_pe, s_pw, s_dlw, s_b,
      out, out + NDOC);
}
